// SelfAttention_77309411617
// MI455X (gfx1250) — compile-verified, matrix operand order checked by reading, not run on hardware
//
#include <hip/hip_runtime.h>
#include <hip/hip_bf16.h>

// ---------------------------------------------------------------------------
// Self-attention (B=4, N=2048, C=256, H=8, Dh=32) with key-padding mask.
// All matmuls via v_wmma_f32_16x16x32_f16 (f16 in, f32 accumulate).
// Flash-attention style; 2x2 register blocking in GEMMs, 32 queries/wave in
// the attention loop so each K/V tile load feeds 4 WMMAs.
// ---------------------------------------------------------------------------

typedef __attribute__((ext_vector_type(16))) _Float16 v16h;
typedef __attribute__((ext_vector_type(8)))  _Float16 v8h;
typedef __attribute__((ext_vector_type(4)))  _Float16 v4h;
typedef __attribute__((ext_vector_type(8)))  float    v8f;

#define BB 4
#define NN 2048
#define CC 256
#define HH 8
#define DH 32
#define MM (BB * NN)          // 8192 rows
#define QKV_N (3 * CC)        // 768
#define SCALE 0.17677669529663687f  // Dh^-0.5

// Load one WMMA 16-bit operand row (16 halves) in the CDNA5 A/B swizzle:
// caller passes p = rowBase + kb where kb = 8*(lane>=16); chunks at +0, +16.
__device__ __forceinline__ v16h ld_row16(const _Float16* p) {
  v8h lo = *reinterpret_cast<const v8h*>(p);
  v8h hi = *reinterpret_cast<const v8h*>(p + 16);
  return __builtin_shufflevector(lo, hi, 0, 1, 2, 3, 4, 5, 6, 7,
                                 8, 9, 10, 11, 12, 13, 14, 15);
}

__device__ __forceinline__ v8f wmma_fp16(v16h a, v16h b, v8f c) {
  // (neg_a, A, neg_b, B, c_mod, C, reuse_a, reuse_b)
  return __builtin_amdgcn_wmma_f32_16x16x32_f16(false, a, false, b,
                                                (short)0, c, false, false);
}

// ---------------------------------------------------------------------------
// Kernel 0: fp32 -> fp16 conversion (4 elements / thread)
// ---------------------------------------------------------------------------
__global__ void cvt_f32_f16_kernel(const float4* __restrict__ src,
                                   _Float16* __restrict__ dst, int n4) {
  int i = blockIdx.x * blockDim.x + threadIdx.x;
  if (i < n4) {
    float4 v = src[i];
    v4h o = {(_Float16)v.x, (_Float16)v.y, (_Float16)v.z, (_Float16)v.w};
    *reinterpret_cast<v4h*>(dst + 4 * i) = o;
  }
}

// ---------------------------------------------------------------------------
// Kernel 1: QKV projection.  out[m, c3] = sum_k x[m,k] * w_qkv[c3,k]
// One wave per 32x32 output tile (2x2 WMMA blocking); scatter into
// q[B,H,N,Dh] (pre-scaled), k[B,H,N,Dh], vT[B,H,Dh,N].
// ---------------------------------------------------------------------------
__global__ void qkv_wmma_kernel(const _Float16* __restrict__ xh,
                                const _Float16* __restrict__ wqh,
                                _Float16* __restrict__ qh,
                                _Float16* __restrict__ kh,
                                _Float16* __restrict__ vt) {
  const int lane   = threadIdx.x & 31;
  const int wave   = threadIdx.x >> 5;
  const int gw     = blockIdx.x * (blockDim.x >> 5) + wave;
  const int tile_m = gw & 255;          // MM/32 = 256
  const int tile_n = gw >> 8;           // 0..23
  const int n  = lane & 15;
  const int kb = (lane >> 4) * 8;
  const int hi = lane >> 4;

  const _Float16* a0p = xh  + ((tile_m * 32 + n) * CC) + kb;
  const _Float16* a1p = a0p + 16 * CC;
  const _Float16* b0p = wqh + ((tile_n * 32 + n) * CC) + kb;
  const _Float16* b1p = b0p + 16 * CC;

  v8f acc00 = {}, acc01 = {}, acc10 = {}, acc11 = {};
#pragma unroll
  for (int kk = 0; kk < CC; kk += 32) {
    v16h a0 = ld_row16(a0p + kk);
    v16h a1 = ld_row16(a1p + kk);
    v16h b0 = ld_row16(b0p + kk);
    v16h b1 = ld_row16(b1p + kk);
    acc00 = wmma_fp16(a0, b0, acc00);
    acc01 = wmma_fp16(a0, b1, acc01);
    acc10 = wmma_fp16(a1, b0, acc10);
    acc11 = wmma_fp16(a1, b1, acc11);
  }

  // All 32 columns of this tile share one (s, head): 32-col tiles align.
  const int col0 = tile_n * 32 + n;          // cols of acc*0
  const int s = col0 >> 8;                   // 0=q 1=k 2=v
  const int h = (col0 >> 5) & 7;
  const int d0 = col0 & 31;                  // n
  const int d1 = d0 + 16;                    // cols of acc*1
#pragma unroll
  for (int mi = 0; mi < 2; ++mi) {
    v8f aL = mi ? acc10 : acc00;             // d = d0
    v8f aR = mi ? acc11 : acc01;             // d = d1
#pragma unroll
    for (int r = 0; r < 8; ++r) {
      int row = tile_m * 32 + mi * 16 + r + 8 * hi;
      int b_  = row >> 11;                   // / NN
      int n_  = row & (NN - 1);
      int bh  = b_ * HH + h;
      float vL = aL[r], vR = aR[r];
      if (s == 0) {
        qh[(bh * NN + n_) * DH + d0] = (_Float16)(vL * SCALE);
        qh[(bh * NN + n_) * DH + d1] = (_Float16)(vR * SCALE);
      } else if (s == 1) {
        kh[(bh * NN + n_) * DH + d0] = (_Float16)vL;
        kh[(bh * NN + n_) * DH + d1] = (_Float16)vR;
      } else {
        vt[(bh * DH + d0) * NN + n_] = (_Float16)vL;
        vt[(bh * DH + d1) * NN + n_] = (_Float16)vR;
      }
    }
  }
}

// ---------------------------------------------------------------------------
// Online-softmax update for one 16-query tile against 32 keys.
// S0/S1: fresh scores (keys lo/hi 16).  Updates running max/sum, rescales
// the O accumulators, and stages exp(S) into the LDS P tile (D layout).
// ---------------------------------------------------------------------------
__device__ __forceinline__ void softmax_update(v8f S0, v8f S1,
                                               bool msk0, bool msk1,
                                               float* Mr, float* Lr,
                                               v8f& Oa, v8f& Ob,
                                               _Float16* Pt,  // [16*32] LDS
                                               int n, int hi) {
#pragma unroll
  for (int r = 0; r < 8; ++r) {
    float s0 = msk0 ? -1e30f : S0[r];
    float s1 = msk1 ? -1e30f : S1[r];
    float t = fmaxf(s0, s1);
    t = fmaxf(t, __shfl_xor(t, 1, 32));
    t = fmaxf(t, __shfl_xor(t, 2, 32));
    t = fmaxf(t, __shfl_xor(t, 4, 32));
    t = fmaxf(t, __shfl_xor(t, 8, 32));   // masks <= 8: stays in 16-lane half
    float mnew  = fmaxf(Mr[r], t);
    float alpha = __expf(Mr[r] - mnew);
    float p0 = __expf(s0 - mnew);
    float p1 = __expf(s1 - mnew);
    float rs = p0 + p1;
    rs += __shfl_xor(rs, 1, 32);
    rs += __shfl_xor(rs, 2, 32);
    rs += __shfl_xor(rs, 4, 32);
    rs += __shfl_xor(rs, 8, 32);
    Lr[r] = Lr[r] * alpha + rs;
    Mr[r] = mnew;
    Oa[r] *= alpha;
    Ob[r] *= alpha;
    int prow = r + 8 * hi;                // D layout: row r+8*hi, col n / 16+n
    Pt[prow * 32 + n]      = (_Float16)p0;
    Pt[prow * 32 + 16 + n] = (_Float16)p1;
  }
}

// ---------------------------------------------------------------------------
// Kernel 2: flash attention.  One wave per (b, h, 32-query tile).
// Per 32-key step: 4 S-WMMAs, online softmax, P via LDS relayout, 4 O-WMMAs.
// K/V tile loads are shared by both query sub-tiles.
// ---------------------------------------------------------------------------
__global__ void attn_wmma_kernel(const _Float16* __restrict__ qh,
                                 const _Float16* __restrict__ kh,
                                 const _Float16* __restrict__ vt,
                                 const int* __restrict__ lens,
                                 _Float16* __restrict__ attnh) {
  __shared__ __align__(16) _Float16 Pt[2][16 * 32];

  const int lane = threadIdx.x & 31;
  const int qt = blockIdx.x & 63;       // N/32 = 64 query tiles
  const int bh = blockIdx.x >> 6;       // b*H + h
  const int b  = bh >> 3;
  const int h  = bh & 7;
  const int q0 = qt * 32;
  const int n  = lane & 15;
  const int kb = (lane >> 4) * 8;
  const int hi = lane >> 4;

  // key_padding_mask may be int64 (jax x64) or int32 (default jax).
  // Lengths are >= 1, so word[1]==0 <=> int64 little-endian layout.
  const bool is64 = (lens[1] == 0);
  const int len = is64 ? lens[2 * b] : lens[b];

  v16h qa0 = ld_row16(qh + (bh * NN + q0 + n) * DH + kb);
  v16h qa1 = ld_row16(qh + (bh * NN + q0 + 16 + n) * DH + kb);

  float Mr0[8], Lr0[8], Mr1[8], Lr1[8];
  v8f O00 = {}, O01 = {}, O10 = {}, O11 = {};   // [qtile][d-half]
#pragma unroll
  for (int r = 0; r < 8; ++r) {
    Mr0[r] = -1e30f; Lr0[r] = 0.0f;
    Mr1[r] = -1e30f; Lr1[r] = 0.0f;
  }

  for (int k0 = 0; k0 < NN; k0 += 32) {
    if (k0 >= len) break;               // uniform per block: safe with WMMA

    v16h kt0 = ld_row16(kh + (bh * NN + k0 + n) * DH + kb);
    v16h kt1 = ld_row16(kh + (bh * NN + k0 + 16 + n) * DH + kb);
    v8f z = {};
    v8f S00 = wmma_fp16(qa0, kt0, z);   // queries q0+0..15,  keys k0+0..15
    v8f S01 = wmma_fp16(qa0, kt1, z);   //                    keys k0+16..31
    v8f S10 = wmma_fp16(qa1, kt0, z);   // queries q0+16..31
    v8f S11 = wmma_fp16(qa1, kt1, z);

    const bool msk0 = (k0 + n) >= len;
    const bool msk1 = (k0 + 16 + n) >= len;

    softmax_update(S00, S01, msk0, msk1, Mr0, Lr0, O00, O01, Pt[0], n, hi);
    softmax_update(S10, S11, msk0, msk1, Mr1, Lr1, O10, O11, Pt[1], n, hi);
    __syncthreads();                    // one wave/block: LDS visibility

    v16h v0  = ld_row16(vt + (bh * DH + n) * NN + k0 + kb);        // d 0..15
    v16h v1  = ld_row16(vt + (bh * DH + 16 + n) * NN + k0 + kb);   // d 16..31
    v16h pa0 = ld_row16(&Pt[0][n * 32 + kb]);                      // A layout
    v16h pa1 = ld_row16(&Pt[1][n * 32 + kb]);
    O00 = wmma_fp16(pa0, v0, O00);
    O01 = wmma_fp16(pa0, v1, O01);
    O10 = wmma_fp16(pa1, v0, O10);
    O11 = wmma_fp16(pa1, v1, O11);
    __syncthreads();
  }

#pragma unroll
  for (int qi = 0; qi < 2; ++qi) {
    const float* Lr = qi ? Lr1 : Lr0;
    v8f Oa = qi ? O10 : O00;
    v8f Ob = qi ? O11 : O01;
#pragma unroll
    for (int r = 0; r < 8; ++r) {
      float inv = (Lr[r] > 0.0f) ? (1.0f / Lr[r]) : 0.0f;
      int row = q0 + qi * 16 + r + 8 * hi;
      int base = (b * NN + row) * CC + h * DH;
      attnh[base + n]      = (_Float16)(Oa[r] * inv);
      attnh[base + 16 + n] = (_Float16)(Ob[r] * inv);
    }
  }
}

// ---------------------------------------------------------------------------
// Kernel 3: output projection.  y[m,c] = sum_k a[m,k]*w_proj[c,k] + b[c]
// One wave per 32x32 output tile (2x2 WMMA blocking).
// ---------------------------------------------------------------------------
__global__ void proj_wmma_kernel(const _Float16* __restrict__ ah,
                                 const _Float16* __restrict__ wph,
                                 const float* __restrict__ bias,
                                 float* __restrict__ y) {
  const int lane   = threadIdx.x & 31;
  const int wave   = threadIdx.x >> 5;
  const int gw     = blockIdx.x * (blockDim.x >> 5) + wave;
  const int tile_m = gw & 255;          // MM/32 = 256
  const int tile_n = gw >> 8;           // 0..7
  const int n  = lane & 15;
  const int kb = (lane >> 4) * 8;
  const int hi = lane >> 4;

  const _Float16* a0p = ah  + ((tile_m * 32 + n) * CC) + kb;
  const _Float16* a1p = a0p + 16 * CC;
  const _Float16* b0p = wph + ((tile_n * 32 + n) * CC) + kb;
  const _Float16* b1p = b0p + 16 * CC;

  v8f acc00 = {}, acc01 = {}, acc10 = {}, acc11 = {};
#pragma unroll
  for (int kk = 0; kk < CC; kk += 32) {
    v16h a0 = ld_row16(a0p + kk);
    v16h a1 = ld_row16(a1p + kk);
    v16h b0 = ld_row16(b0p + kk);
    v16h b1 = ld_row16(b1p + kk);
    acc00 = wmma_fp16(a0, b0, acc00);
    acc01 = wmma_fp16(a0, b1, acc01);
    acc10 = wmma_fp16(a1, b0, acc10);
    acc11 = wmma_fp16(a1, b1, acc11);
  }

  const int col0 = tile_n * 32 + n;
  const int col1 = col0 + 16;
  const float bv0 = bias[col0];
  const float bv1 = bias[col1];
#pragma unroll
  for (int mi = 0; mi < 2; ++mi) {
    v8f aL = mi ? acc10 : acc00;
    v8f aR = mi ? acc11 : acc01;
#pragma unroll
    for (int r = 0; r < 8; ++r) {
      int row = tile_m * 32 + mi * 16 + r + 8 * hi;
      y[row * CC + col0] = aL[r] + bv0;
      y[row * CC + col1] = aR[r] + bv1;
    }
  }
}

// ---------------------------------------------------------------------------
// Launch
// ---------------------------------------------------------------------------
extern "C" void kernel_launch(void* const* d_in, const int* in_sizes, int n_in,
                              void* d_out, int out_size, void* d_ws, size_t ws_size,
                              hipStream_t stream) {
  const float* x      = (const float*)d_in[0];
  const int*   lens   = (const int*)d_in[1];   // int32 or int64 (auto-detected)
  const float* w_qkv  = (const float*)d_in[2];
  const float* w_proj = (const float*)d_in[3];
  const float* b_proj = (const float*)d_in[4];
  float* out = (float*)d_out;

  // workspace layout (fp16 element offsets)
  const size_t E = (size_t)MM * CC;     // 2,097,152
  _Float16* ws    = (_Float16*)d_ws;
  _Float16* xh    = ws;                 // [M, C]
  _Float16* qh    = ws + 1 * E;         // [B,H,N,Dh] (pre-scaled)
  _Float16* kh    = ws + 2 * E;         // [B,H,N,Dh]
  _Float16* vt    = ws + 3 * E;         // [B,H,Dh,N]
  _Float16* attnh = ws + 4 * E;         // [B,N,C]
  _Float16* wqh   = ws + 5 * E;         // [3C, C]
  _Float16* wph   = wqh + (size_t)QKV_N * CC;  // [C, C]

  const int nx4  = (MM * CC) / 4;
  const int nwq4 = (QKV_N * CC) / 4;
  const int nwp4 = (CC * CC) / 4;
  cvt_f32_f16_kernel<<<(nx4  + 255) / 256, 256, 0, stream>>>((const float4*)x, xh, nx4);
  cvt_f32_f16_kernel<<<(nwq4 + 255) / 256, 256, 0, stream>>>((const float4*)w_qkv, wqh, nwq4);
  cvt_f32_f16_kernel<<<(nwp4 + 255) / 256, 256, 0, stream>>>((const float4*)w_proj, wph, nwp4);

  // QKV: (8192/32) * (768/32) = 256*24 = 6144 wave-tiles, 8 waves / block
  qkv_wmma_kernel<<<6144 / 8, 256, 0, stream>>>(xh, wqh, qh, kh, vt);

  // Attention: B*H*(N/32) = 2048 blocks of one wave
  attn_wmma_kernel<<<BB * HH * (NN / 32), 32, 0, stream>>>(qh, kh, vt, lens, attnh);

  // Projection: (8192/32) * (256/32) = 256*8 = 2048 wave-tiles, 8 waves / block
  proj_wmma_kernel<<<2048 / 8, 256, 0, stream>>>(attnh, wph, b_proj, out);
}
